// SAGEELayer_33200097198873
// MI455X (gfx1250) — matrix operand order repaired and checked by reading, not compiled, not run on hardware
//
#include <hip/hip_runtime.h>
#include <hip/hip_bf16.h>

// CDNA5 / gfx1250: wave32, WMMA 16x16x32 bf16 with fp32 accumulate.
typedef __attribute__((ext_vector_type(16))) __bf16 v16bf;
typedef __attribute__((ext_vector_type(8)))  float  v8f;

#define N_NODES 50000
#define N_EDGES 800000
#define NDIM_IN 64
#define EDIMS   64
#define NDIM_OUT 128
#define K_MSG (NDIM_IN + EDIMS)      // 128
#define K_APP (NDIM_IN + NDIM_OUT)   // 192
#define TILE_E 256                   // edges per block tile (32 per wave)

// padded LDS strides (in 16-bit elements): multiple of 8 (16B align for b128),
// chosen so row/column bases hit distinct bank groups (64 banks x 4B).
#define WSTRIDE1 136   // K_MSG + 8  -> 272B rows
#define ASTRIDE1 136
#define WSTRIDE2 200   // K_APP + 8  -> 400B rows
#define ASTRIDE2 200

__device__ __forceinline__ unsigned short f2bf(float f) {
  unsigned int u = __float_as_uint(f);
  u += 0x7FFFu + ((u >> 16) & 1u);   // round-to-nearest-even
  return (unsigned short)(u >> 16);
}

union AFrag { v16bf v; uint4 q[2]; };

__global__ void zero_kernel(float4* __restrict__ p, int n4) {
  int i = blockIdx.x * blockDim.x + threadIdx.x;
  if (i < n4) p[i] = make_float4(0.f, 0.f, 0.f, 0.f);
}

// ---------------------------------------------------------------------------
// Kernel 1: per-edge message GEMM + ReLU + scatter-add into h_neigh.
// Block = 256 threads = 8 waves; each wave owns 32 edges (two 16-row M-tiles)
// of a 256-edge block tile, so every B fragment feeds two WMMAs.
// ---------------------------------------------------------------------------
__global__ __launch_bounds__(256) void edge_msg_kernel(
    const float* __restrict__ nfeats, const float* __restrict__ efeats,
    const float* __restrict__ Wm, const float* __restrict__ bm,
    const long long* __restrict__ src, const long long* __restrict__ dst,
    float* __restrict__ hneigh)
{
  __shared__ unsigned short wLds[NDIM_OUT * WSTRIDE1]; // W_msg^T: [n][k] bf16
  __shared__ unsigned short aLds[8][32 * ASTRIDE1];    // per-wave 32-row A tiles
  __shared__ int sTile[TILE_E];
  __shared__ int dTile[TILE_E];

  const int tid  = threadIdx.x;
  const int lane = tid & 31;
  const int w    = tid >> 5;        // wave id 0..7
  const int kh   = lane >> 4;       // lane half (K-half selector)
  const int ln   = lane & 15;       // row / column within 16

  // Stage W_msg (global [k][n] row-major) transposed into LDS as bf16.
  for (int idx = tid; idx < K_MSG * NDIM_OUT; idx += 256) {
    const int k = idx >> 7;          // NDIM_OUT == 128
    const int n = idx & 127;
    wLds[n * WSTRIDE1 + k] = f2bf(Wm[idx]);
  }

  float biasReg[8];
#pragma unroll
  for (int nt = 0; nt < 8; ++nt) biasReg[nt] = bm[nt * 16 + ln];

  const int numTiles = N_EDGES / TILE_E;   // 3125, exact
  for (int tile = blockIdx.x; tile < numTiles; tile += gridDim.x) {
    const int eBase = tile * TILE_E;

    __syncthreads();   // protect sTile/dTile/aLds reuse (and first-use of wLds)
    sTile[tid] = (int)src[eBase + tid];
    dTile[tid] = (int)dst[eBase + tid];
    // prefetch next tile's edge features (global_prefetch_b8)
    {
      const int tNext = tile + gridDim.x;
      if (tNext < numTiles)
        __builtin_prefetch(efeats + (unsigned)tNext * (TILE_E * EDIMS) + tid * 64, 0, 1);
    }
    __syncthreads();

    // Stage A tile: 32 edges x 128 K as bf16. Lanes 0-15 gather nfeats[src],
    // lanes 16-31 stream efeats — each row is one fully coalesced 128-float read.
    const int weBase = eBase + w * 32;
    unsigned short* aRow = aLds[w];
#pragma unroll 4
    for (int r = 0; r < 32; ++r) {
      const float4 f = (lane < 16)
        ? *(const float4*)(nfeats + (unsigned)sTile[w * 32 + r] * NDIM_IN + lane * 4)
        : *(const float4*)(efeats + (unsigned)(weBase + r) * EDIMS + (lane - 16) * 4);
      unsigned short* p = aRow + r * ASTRIDE1 + lane * 4;
      p[0] = f2bf(f.x); p[1] = f2bf(f.y); p[2] = f2bf(f.z); p[3] = f2bf(f.w);
    }
    __syncthreads();

    // A fragments per ISA 16-bit A 16x32 layout:
    // lane half kh: K = 32c + 8*kh + [0..7]  and  32c + 16 + 8*kh + [0..7]
    AFrag a0[4], a1[4];
    const unsigned short* row0 = aRow + ln * ASTRIDE1;
    const unsigned short* row1 = aRow + (16 + ln) * ASTRIDE1;
#pragma unroll
    for (int c = 0; c < 4; ++c) {
      a0[c].q[0] = *(const uint4*)(row0 + c * 32 + kh * 8);
      a0[c].q[1] = *(const uint4*)(row0 + c * 32 + 16 + kh * 8);
      a1[c].q[0] = *(const uint4*)(row1 + c * 32 + kh * 8);
      a1[c].q[1] = *(const uint4*)(row1 + c * 32 + 16 + kh * 8);
    }

#pragma unroll
    for (int nt = 0; nt < 8; ++nt) {
      v8f acc0 = {};
      v8f acc1 = {};
      // B fragment: column j = nt*16+ln, K = 32c + 8*kh + [0..7] and 32c + 16 + 8*kh + [0..7]
      const unsigned short* bcol = wLds + (nt * 16 + ln) * WSTRIDE1;
#pragma unroll
      for (int c = 0; c < 4; ++c) {
        AFrag b;
        b.q[0] = *(const uint4*)(bcol + c * 32 + kh * 8);
        b.q[1] = *(const uint4*)(bcol + c * 32 + 16 + kh * 8);
        acc0 = __builtin_amdgcn_wmma_f32_16x16x32_bf16(
            false, a0[c].v, false, b.v, (short)0, acc0, false, false);
        acc1 = __builtin_amdgcn_wmma_f32_16x16x32_bf16(
            false, a1[c].v, false, b.v, (short)0, acc1, false, false);
      }
      const int col = nt * 16 + ln;
#pragma unroll
      for (int v = 0; v < 8; ++v) {
        const int r = v + kh * 8;                   // C layout: M = v + 8*kh
        // 32-bit element offsets -> SADDR + voffset global atomics
        const unsigned off0 = (unsigned)dTile[w * 32 + r] * NDIM_OUT + col;
        const unsigned off1 = (unsigned)dTile[w * 32 + 16 + r] * NDIM_OUT + col;
        const float v0 = fmaxf(acc0[v] + biasReg[nt], 0.0f);
        const float v1 = fmaxf(acc1[v] + biasReg[nt], 0.0f);
        __hip_atomic_fetch_add(hneigh + off0, v0,
                               __ATOMIC_RELAXED, __HIP_MEMORY_SCOPE_AGENT);
        __hip_atomic_fetch_add(hneigh + off1, v1,
                               __ATOMIC_RELAXED, __HIP_MEMORY_SCOPE_AGENT);
      }
    }
  }
}

// ---------------------------------------------------------------------------
// Kernel 2: out = relu(concat(nfeats, h_neigh) @ W_apply + b_apply).
// Block = 128 threads = 4 waves; 64 nodes per block.
// ---------------------------------------------------------------------------
__global__ __launch_bounds__(128) void apply_kernel(
    const float* __restrict__ nfeats, const float* __restrict__ hneigh,
    const float* __restrict__ Wa, const float* __restrict__ ba,
    float* __restrict__ out)
{
  __shared__ unsigned short wLds[NDIM_OUT * WSTRIDE2]; // W_apply^T: [n][k] bf16
  __shared__ unsigned short aLds[4][16 * ASTRIDE2];

  const int tid  = threadIdx.x;
  const int lane = tid & 31;
  const int w    = tid >> 5;    // 0..3
  const int kh   = lane >> 4;
  const int ln   = lane & 15;

  for (int idx = tid; idx < K_APP * NDIM_OUT; idx += 128) {
    const int k = idx >> 7;
    const int n = idx & 127;
    wLds[n * WSTRIDE2 + k] = f2bf(Wa[idx]);
  }

  float biasReg[8];
#pragma unroll
  for (int nt = 0; nt < 8; ++nt) biasReg[nt] = ba[nt * 16 + ln];

  const int nodeBase = blockIdx.x * 64 + w * 16;
  unsigned short* aRow = aLds[w];

  // Stage A: 16 nodes x 192 K (k<64: nfeats, k>=64: h_neigh), bf16.
#pragma unroll 2
  for (int r = 0; r < 16; ++r) {
    const int node = nodeBase + r;
    const bool ok = node < N_NODES;
    if (lane < 16) {
      const float4 f = ok
        ? *(const float4*)(nfeats + (unsigned)node * NDIM_IN + lane * 4)
        : make_float4(0.f, 0.f, 0.f, 0.f);
      unsigned short* p = aRow + r * ASTRIDE2 + lane * 4;
      p[0] = f2bf(f.x); p[1] = f2bf(f.y); p[2] = f2bf(f.z); p[3] = f2bf(f.w);
    }
    {
      const float4 f = ok
        ? *(const float4*)(hneigh + (unsigned)node * NDIM_OUT + lane * 4)
        : make_float4(0.f, 0.f, 0.f, 0.f);
      unsigned short* p = aRow + r * ASTRIDE2 + 64 + lane * 4;
      p[0] = f2bf(f.x); p[1] = f2bf(f.y); p[2] = f2bf(f.z); p[3] = f2bf(f.w);
    }
  }
  __syncthreads();

  AFrag a[6];
  const unsigned short* myRow = aRow + ln * ASTRIDE2;
#pragma unroll
  for (int c = 0; c < 6; ++c) {
    a[c].q[0] = *(const uint4*)(myRow + c * 32 + kh * 8);
    a[c].q[1] = *(const uint4*)(myRow + c * 32 + 16 + kh * 8);
  }

#pragma unroll
  for (int nt = 0; nt < 8; ++nt) {
    v8f acc = {};
    const unsigned short* bcol = wLds + (nt * 16 + ln) * WSTRIDE2;
#pragma unroll
    for (int c = 0; c < 6; ++c) {
      AFrag b;
      b.q[0] = *(const uint4*)(bcol + c * 32 + kh * 8);
      b.q[1] = *(const uint4*)(bcol + c * 32 + 16 + kh * 8);
      acc = __builtin_amdgcn_wmma_f32_16x16x32_bf16(
          false, a[c].v, false, b.v, (short)0, acc, false, false);
    }
    const int col = nt * 16 + ln;
#pragma unroll
    for (int v = 0; v < 8; ++v) {
      const int node = nodeBase + v + kh * 8;
      if (node < N_NODES)
        out[(unsigned)node * NDIM_OUT + col] = fmaxf(acc[v] + biasReg[nt], 0.0f);
    }
  }
}

// ---------------------------------------------------------------------------
extern "C" void kernel_launch(void* const* d_in, const int* in_sizes, int n_in,
                              void* d_out, int out_size, void* d_ws, size_t ws_size,
                              hipStream_t stream) {
  const float*     nfeats = (const float*)d_in[0];
  const float*     efeats = (const float*)d_in[1];
  const float*     Wm     = (const float*)d_in[2];
  const float*     bm     = (const float*)d_in[3];
  const float*     Wa     = (const float*)d_in[4];
  const float*     ba     = (const float*)d_in[5];
  const long long* src    = (const long long*)d_in[6];  // jnp.int64
  const long long* dst    = (const long long*)d_in[7];  // jnp.int64
  float*           outp   = (float*)d_out;
  float*           hneigh = (float*)d_ws;                // 50000*128 f32 = 25.6 MB

  (void)in_sizes; (void)n_in; (void)out_size; (void)ws_size;

  const int n4 = (N_NODES * NDIM_OUT) / 4;  // 1,600,000 float4
  zero_kernel<<<(n4 + 255) / 256, 256, 0, stream>>>((float4*)hneigh, n4);

  edge_msg_kernel<<<640, 256, 0, stream>>>(nfeats, efeats, Wm, bm, src, dst, hneigh);

  apply_kernel<<<(N_NODES + 63) / 64, 128, 0, stream>>>(nfeats, hneigh, Wa, ba, outp);
}
